// PointNetPlusPlusSegmentation_20650202759618
// MI455X (gfx1250) — matrix operand order repaired and checked by reading, not compiled, not run on hardware
//
#include <hip/hip_runtime.h>
#include <cmath>

typedef __attribute__((ext_vector_type(16))) _Float16 v16h;
typedef __attribute__((ext_vector_type(8)))  _Float16 h8v;
typedef __attribute__((ext_vector_type(8)))  float    v8f;

#define BN_SCALE 0.9999950000374997f   // 1/sqrt(1 + 1e-5)

// ---------------------------------------------------------------------------
// CDNA5 async memory->LDS copy (ASYNCcnt-tracked), ISA 08_async_tensor.
// LDS destination = low 32 bits of the generic shared address (LDS aperture
// maps addr[31:0] to the wave's LDS allocation, ISA 10.2).
// ---------------------------------------------------------------------------
__device__ __forceinline__ void async_copy16(void* lds, const void* gsrc) {
  unsigned l = (unsigned)(unsigned long long)(size_t)lds;
  asm volatile("global_load_async_to_lds_b128 %0, %1, off"
               :: "v"(l), "v"(gsrc) : "memory");
}
__device__ __forceinline__ void wait_async0() {
  asm volatile("s_wait_asynccnt 0x0" ::: "memory");
}
__device__ __forceinline__ void wait_async3() {
  asm volatile("s_wait_asynccnt 0x3" ::: "memory");
}

// ---------------------------------------------------------------------------
// transpose x (B,3,N) -> xyz (B,N,3)
// ---------------------------------------------------------------------------
__global__ void k_transpose(const float* __restrict__ x, float* __restrict__ xyz,
                            int B, int N) {
  int t = blockIdx.x * blockDim.x + threadIdx.x;
  if (t >= B * N) return;
  int b = t / N, n = t % N;
  xyz[(size_t)t * 3 + 0] = x[((size_t)b * 3 + 0) * N + n];
  xyz[(size_t)t * 3 + 1] = x[((size_t)b * 3 + 1) * N + n];
  xyz[(size_t)t * 3 + 2] = x[((size_t)b * 3 + 2) * N + n];
}

// ---------------------------------------------------------------------------
// farthest point sampling: one block per batch, sequential npoint iterations,
// block-wide argmax with first-index tie-break (matches jnp.argmax).
// ---------------------------------------------------------------------------
__global__ __launch_bounds__(256) void k_fps(const float* __restrict__ xyz,
                                             int N, int npoint,
                                             int* __restrict__ fi) {
  __shared__ float dist[4096];
  __shared__ float sval[256];
  __shared__ int   sidx[256];
  int b = blockIdx.x;
  const float* base = xyz + (size_t)b * N * 3;
  for (int n = threadIdx.x; n < N; n += 256) dist[n] = 1e10f;
  __syncthreads();
  int far = 0;
  for (int i = 0; i < npoint; ++i) {
    if (threadIdx.x == 0) fi[(size_t)b * npoint + i] = far;
    float fx = base[far * 3 + 0], fy = base[far * 3 + 1], fz = base[far * 3 + 2];
    float best = -1.0f; int bidx = 0;
    for (int n = threadIdx.x; n < N; n += 256) {
      float dx = base[n * 3 + 0] - fx;
      float dy = base[n * 3 + 1] - fy;
      float dz = base[n * 3 + 2] - fz;
      float d = dx * dx + dy * dy + dz * dz;
      float dn = fminf(dist[n], d);
      dist[n] = dn;
      if (dn > best) { best = dn; bidx = n; }
    }
    sval[threadIdx.x] = best; sidx[threadIdx.x] = bidx;
    __syncthreads();
    for (int off = 128; off > 0; off >>= 1) {
      if ((int)threadIdx.x < off) {
        float v2 = sval[threadIdx.x + off]; int i2 = sidx[threadIdx.x + off];
        if (v2 > sval[threadIdx.x] ||
            (v2 == sval[threadIdx.x] && i2 < sidx[threadIdx.x])) {
          sval[threadIdx.x] = v2; sidx[threadIdx.x] = i2;
        }
      }
      __syncthreads();
    }
    far = sidx[0];
    __syncthreads();
  }
}

// gather 3-vectors: out[b][s] = xyz[b][fi[b][s]]
__global__ void k_gather3(const float* __restrict__ xyz, const int* __restrict__ fi,
                          int N, int S, int total, float* __restrict__ out) {
  int t = blockIdx.x * blockDim.x + threadIdx.x;
  if (t >= total) return;
  int b = t / S;
  int n = fi[t];
  const float* p = xyz + ((size_t)b * N + n) * 3;
  out[(size_t)t * 3 + 0] = p[0];
  out[(size_t)t * 3 + 1] = p[1];
  out[(size_t)t * 3 + 2] = p[2];
}

// ---------------------------------------------------------------------------
// ball query: in-order scan over N == sorted indices, fill with first hit
// ---------------------------------------------------------------------------
__global__ void k_query_ball(const float* __restrict__ xyz,
                             const float* __restrict__ new_xyz,
                             int N, int S, int nsample, float r2, int total,
                             int* __restrict__ gi) {
  int t = blockIdx.x * blockDim.x + threadIdx.x;
  if (t >= total) return;
  int b = t / S;
  const float* q = new_xyz + (size_t)t * 3;
  float qx = q[0], qy = q[1], qz = q[2];
  const float* base = xyz + (size_t)b * N * 3;
  int* out = gi + (size_t)t * nsample;
  int cnt = 0, first = 0;
  for (int n = 0; n < N && cnt < nsample; ++n) {
    float dx = base[n * 3 + 0] - qx;
    float dy = base[n * 3 + 1] - qy;
    float dz = base[n * 3 + 2] - qz;
    float d = dx * dx + dy * dy + dz * dz;
    if (d <= r2) {
      if (cnt == 0) first = n;
      out[cnt++] = n;
    }
  }
  for (; cnt < nsample; ++cnt) out[cnt] = first;
}

// ---------------------------------------------------------------------------
// build grouped f16 GEMM input (zero-padded to Cinp): [rel_xyz | points[gi]]
// ---------------------------------------------------------------------------
__global__ void k_group(const float* __restrict__ xyz, const float* __restrict__ new_xyz,
                        const int* __restrict__ gi, const _Float16* __restrict__ points,
                        int C, int N, int S, int J, int total, int Cinp,
                        _Float16* __restrict__ X) {
  int t = blockIdx.x * blockDim.x + threadIdx.x;
  if (t >= total) return;
  int b = t / (S * J);
  int s = (t / J) % S;
  int n = gi[t];
  _Float16* row = X + (size_t)t * Cinp;
  const float* p = xyz + ((size_t)b * N + n) * 3;
  const float* q = new_xyz + ((size_t)b * S + s) * 3;
  row[0] = (_Float16)(p[0] - q[0]);
  row[1] = (_Float16)(p[1] - q[1]);
  row[2] = (_Float16)(p[2] - q[2]);
  if (C > 0) {
    const _Float16* f = points + ((size_t)b * N + n) * C;
    for (int c = 0; c < C; ++c) row[3 + c] = f[c];
  }
  for (int c = 3 + C; c < Cinp; ++c) row[c] = (_Float16)0.f;
}

// sa3 "group all": rows = b*S + j, channels = [l2x(3) | l2p(256)] padded to Cinp
__global__ void k_group_all(const float* __restrict__ l2x, const _Float16* __restrict__ l2p,
                            int rows, int Cinp, _Float16* __restrict__ X) {
  int t = blockIdx.x * blockDim.x + threadIdx.x;
  if (t >= rows) return;
  _Float16* row = X + (size_t)t * Cinp;
  const float* p = l2x + (size_t)t * 3;
  row[0] = (_Float16)p[0];
  row[1] = (_Float16)p[1];
  row[2] = (_Float16)p[2];
  const _Float16* f = l2p + (size_t)t * 256;
  for (int c = 0; c < 256; ++c) row[3 + c] = f[c];
  for (int c = 259; c < Cinp; ++c) row[c] = (_Float16)0.f;
}

// fp3 input: rows = b*128+n; [l2p(256) | broadcast l3p(1024)]  (1280 = 40*32)
__global__ void k_concat_fp3(const _Float16* __restrict__ l2p,
                             const _Float16* __restrict__ l3p,
                             int rows, _Float16* __restrict__ X) {
  int t = blockIdx.x * blockDim.x + threadIdx.x;
  int total = rows * 1280;
  if (t >= total) return;
  int row = t / 1280, c = t % 1280;
  int b = row >> 7;
  X[t] = (c < 256) ? l2p[(size_t)row * 256 + c]
                   : l3p[(size_t)b * 1024 + (c - 256)];
}

// ---------------------------------------------------------------------------
// 3-nearest-neighbor + inverse-distance weights (feature propagation)
// ---------------------------------------------------------------------------
__global__ void k_knn3(const float* __restrict__ xyz1, const float* __restrict__ xyz2,
                       int total, int Nper, int S2,
                       int* __restrict__ kidx, float* __restrict__ kw) {
  int t = blockIdx.x * blockDim.x + threadIdx.x;
  if (t >= total) return;
  int b = t / Nper;
  const float* q = xyz1 + (size_t)t * 3;
  float qx = q[0], qy = q[1], qz = q[2];
  const float* base = xyz2 + (size_t)b * S2 * 3;
  float d0 = 1e30f, d1 = 1e30f, d2 = 1e30f;
  int i0 = 0, i1 = 0, i2 = 0;
  for (int s = 0; s < S2; ++s) {
    float dx = base[s * 3 + 0] - qx;
    float dy = base[s * 3 + 1] - qy;
    float dz = base[s * 3 + 2] - qz;
    float d = dx * dx + dy * dy + dz * dz;
    if (d < d0)      { d2 = d1; i2 = i1; d1 = d0; i1 = i0; d0 = d; i0 = s; }
    else if (d < d1) { d2 = d1; i2 = i1; d1 = d;  i1 = s; }
    else if (d < d2) { d2 = d;  i2 = s; }
  }
  float r0 = 1.0f / (d0 + 1e-8f);
  float r1 = 1.0f / (d1 + 1e-8f);
  float r2 = 1.0f / (d2 + 1e-8f);
  float inv = 1.0f / (r0 + r1 + r2);
  kidx[(size_t)t * 3 + 0] = i0; kw[(size_t)t * 3 + 0] = r0 * inv;
  kidx[(size_t)t * 3 + 1] = i1; kw[(size_t)t * 3 + 1] = r1 * inv;
  kidx[(size_t)t * 3 + 2] = i2; kw[(size_t)t * 3 + 2] = r2 * inv;
}

// build FP input: [points1(C1) | sum_k w_k * points2[idx_k] (C2)]  (C1+C2 % 32 == 0)
__global__ void k_interp_concat(const _Float16* __restrict__ p1, int C1,
                                const _Float16* __restrict__ p2, int C2,
                                const int* __restrict__ kidx, const float* __restrict__ kw,
                                int rows, int S2, _Float16* __restrict__ X) {
  int Cin = C1 + C2;
  long long t = (long long)blockIdx.x * blockDim.x + threadIdx.x;
  if (t >= (long long)rows * Cin) return;
  int row = (int)(t / Cin);
  int c = (int)(t % Cin);
  if (c < C1) { X[t] = p1[(size_t)row * C1 + c]; return; }
  int b = row / (rows / 16);
  int c2 = c - C1;
  const int* id = kidx + (size_t)row * 3;
  const float* w = kw + (size_t)row * 3;
  float v = 0.f;
  for (int k = 0; k < 3; ++k)
    v += w[k] * (float)p2[((size_t)b * S2 + id[k]) * C2 + c2];
  X[t] = (_Float16)v;
}

// max over group dimension J
__global__ void k_maxpool(const _Float16* __restrict__ Y, int G, int J, int C,
                          _Float16* __restrict__ out) {
  long long t = (long long)blockIdx.x * blockDim.x + threadIdx.x;
  if (t >= (long long)G * C) return;
  int g = (int)(t / C), c = (int)(t % C);
  float m = -1e30f;
  for (int j = 0; j < J; ++j) {
    float v = (float)Y[((size_t)g * J + j) * C + c];
    m = v > m ? v : m;
  }
  out[t] = (_Float16)m;
}

// ---------------------------------------------------------------------------
// weight prep: fp32 (Cout,Cin) -> f16 (Coutp,Cinp), zero padded
// ---------------------------------------------------------------------------
__global__ void k_convert_w(const float* __restrict__ W, int Cin, int Cout,
                            int Cinp, int Coutp, _Float16* __restrict__ Wh) {
  long long t = (long long)blockIdx.x * blockDim.x + threadIdx.x;
  if (t >= (long long)Coutp * Cinp) return;
  int o = (int)(t / Cinp), k = (int)(t % Cinp);
  Wh[t] = (o < Cout && k < Cin) ? (_Float16)W[(size_t)o * Cin + k] : (_Float16)0.f;
}

// ---------------------------------------------------------------------------
// WMMA GEMM + fused bias/BN/ReLU epilogue, async-to-LDS double buffered.
//   Y[r][o] = act( g[o] * ((X[r,:]·W[o,:] + b[o]) * BN_SCALE) + bb[o] )
// X: (R, Cinp) f16;  Wh: (Coutp, Cinp) f16 zero-padded.
// R % 64 == 0, Cinp % 32 == 0, Coutp % 32 == 0 guaranteed by host.
// 128 threads = 4 waves; block tile 64(M) x 32(N); each wave: 16x32 (2 wmma).
// Staging: global_load_async_to_lds_b128 (ASYNCcnt), prefetch next K-chunk
// into the other LDS buffer while WMMAs consume the current one.
// ---------------------------------------------------------------------------
__global__ __launch_bounds__(128) void k_gemm(
    const _Float16* __restrict__ X, const _Float16* __restrict__ Wh,
    const float* __restrict__ bvec, const float* __restrict__ g,
    const float* __restrict__ bb,
    _Float16* __restrict__ Yh, float* __restrict__ Yf,
    int R, int Cinp, int Cout, int bnrelu) {
  __shared__ _Float16 Xs[2][64][40];   // 80B row stride: 16B-aligned
  __shared__ _Float16 Ws[2][32][40];
  int rowBase = blockIdx.x * 64;
  int colBase = blockIdx.y * 32;
  int tid  = threadIdx.x;
  int w    = tid >> 5;   // wave id 0..3 (wave32)
  int lane = tid & 31;
  int nk   = Cinp >> 5;

  // per-thread staging assignment: W tile 32x32 = 128 chunks (1/thread),
  // X tile 64x32 = 256 chunks (2/thread: rows i and i+32, same chunk col)
  int wi = tid >> 2, wc = tid & 3;
  const _Float16* wsrc  = &Wh[(size_t)(colBase + wi) * Cinp + wc * 8];
  const _Float16* xsrc0 = &X[(size_t)(rowBase + wi) * Cinp + wc * 8];
  const _Float16* xsrc1 = xsrc0 + (size_t)32 * Cinp;

  auto issue = [&](int kc, int s) {
    size_t koff = (size_t)kc * 32;
    async_copy16(&Ws[s][wi][wc * 8],      wsrc  + koff);
    async_copy16(&Xs[s][wi][wc * 8],      xsrc0 + koff);
    async_copy16(&Xs[s][32 + wi][wc * 8], xsrc1 + koff);
  };

  v8f acc0 = {}, acc1 = {};
  issue(0, 0);
  for (int kc = 0; kc < nk; ++kc) {
    int cur = kc & 1;
    if (kc + 1 < nk) {          // prefetch next chunk into other buffer
      issue(kc + 1, cur ^ 1);
      wait_async3();            // current chunk's 3 ops complete (in-order)
    } else {
      wait_async0();
    }
    __syncthreads();
    // A fragment: M = lane&15 ; K = {kbA..kbA+7, kbA+16..kbA+23}, kbA = 8*(lane>=16)
    // B fragment: N = lane&15 ; K = {kbB..kbB+7, kbB+16..kbB+23}, kbB = 8*(lane>=16)
    int m   = lane & 15;
    int kbA = (lane >> 4) << 3;
    int kbB = (lane >> 4) << 3;
    h8v alo  = *(const h8v*)&Xs[cur][w * 16 + m][kbA];
    h8v ahi  = *(const h8v*)&Xs[cur][w * 16 + m][kbA + 16];
    h8v b0lo = *(const h8v*)&Ws[cur][m][kbB];
    h8v b0hi = *(const h8v*)&Ws[cur][m][kbB + 16];
    h8v b1lo = *(const h8v*)&Ws[cur][16 + m][kbB];
    h8v b1hi = *(const h8v*)&Ws[cur][16 + m][kbB + 16];
    v16h a, b0, b1;
#pragma unroll
    for (int i = 0; i < 8; ++i) {
      a[i] = alo[i];  a[8 + i] = ahi[i];
      b0[i] = b0lo[i]; b0[8 + i] = b0hi[i];
      b1[i] = b1lo[i]; b1[8 + i] = b1hi[i];
    }
    acc0 = __builtin_amdgcn_wmma_f32_16x16x32_f16(false, a, false, b0,
                                                  (short)0, acc0, false, false);
    acc1 = __builtin_amdgcn_wmma_f32_16x16x32_f16(false, a, false, b1,
                                                  (short)0, acc1, false, false);
    __syncthreads();            // all waves done reading buffer `cur`
  }
  // epilogue: C/D layout — VGPR i: M = i + 8*(lane>=16), N = lane&15
  int n  = lane & 15;
  int r0 = rowBase + w * 16 + ((lane >> 4) << 3);
#pragma unroll
  for (int t2 = 0; t2 < 2; ++t2) {
    v8f acc = t2 ? acc1 : acc0;
    int o = colBase + t2 * 16 + n;
    if (o < Cout) {
      float bo  = bvec[o];
      float go  = bnrelu ? g[o]  : 0.f;
      float bbo = bnrelu ? bb[o] : 0.f;
#pragma unroll
      for (int i = 0; i < 8; ++i) {
        int r = r0 + i;
        float v = acc[i] + bo;
        if (bnrelu) {
          v = go * (v * BN_SCALE) + bbo;
          v = v > 0.f ? v : 0.f;
          Yh[(size_t)r * Cout + o] = (_Float16)v;
        } else {
          Yf[(size_t)r * Cout + o] = v;
        }
      }
    }
  }
}

// log_softmax over the class dimension
__global__ void k_logsoftmax(const float* __restrict__ logits, float* __restrict__ out,
                             int rows, int C) {
  int t = blockIdx.x * blockDim.x + threadIdx.x;
  if (t >= rows) return;
  const float* l = logits + (size_t)t * C;
  float m = -1e30f;
  for (int c = 0; c < C; ++c) m = fmaxf(m, l[c]);
  float sum = 0.f;
  for (int c = 0; c < C; ++c) sum += expf(l[c] - m);
  float lse = m + logf(sum);
  for (int c = 0; c < C; ++c) out[(size_t)t * C + c] = l[c] - lse;
}

// ---------------------------------------------------------------------------
// host orchestration
// ---------------------------------------------------------------------------
extern "C" void kernel_launch(void* const* d_in, const int* in_sizes, int n_in,
                              void* d_out, int out_size, void* d_ws, size_t ws_size,
                              hipStream_t stream) {
  (void)in_sizes; (void)n_in; (void)out_size; (void)ws_size;
  const int B = 16, N0 = 4096;
  const int S1 = 512, J1 = 32;
  const int S2 = 128, J2 = 64;

  const float* x = (const float*)d_in[0];

  struct Layer { const float *W, *b, *bb, *g; int cin, cout, cinp, coutp; };
  auto mk = [&](int base, int cin, int cout) {
    Layer L;
    L.W  = (const float*)d_in[base + 0];   // 'W'
    L.b  = (const float*)d_in[base + 1];   // 'b'
    L.bb = (const float*)d_in[base + 2];   // 'bb'
    L.g  = (const float*)d_in[base + 3];   // 'g'
    L.cin = cin; L.cout = cout;
    L.cinp  = (cin  + 31) & ~31;
    L.coutp = (cout + 31) & ~31;
    return L;
  };
  // flatten order: x, then params dict with sorted keys:
  // fp1(3), fp2(2), fp3(2), head1, head2, sa1(3), sa2(3), sa3(3)
  Layer fp1a = mk(1, 128, 128),  fp1b = mk(5, 128, 128),  fp1c = mk(9, 128, 128);
  Layer fp2a = mk(13, 384, 256), fp2b = mk(17, 256, 128);
  Layer fp3a = mk(21, 1280, 256), fp3b = mk(25, 256, 256);
  Layer head1 = mk(29, 128, 128);
  Layer head2; head2.W = (const float*)d_in[33]; head2.b = (const float*)d_in[34];
  head2.bb = nullptr; head2.g = nullptr;
  head2.cin = 128; head2.cout = 50; head2.cinp = 128; head2.coutp = 64;
  Layer sa1a = mk(35, 3, 64),    sa1b = mk(39, 64, 64),   sa1c = mk(43, 64, 128);
  Layer sa2a = mk(47, 131, 128), sa2b = mk(51, 128, 128), sa2c = mk(55, 128, 256);
  Layer sa3a = mk(59, 259, 256), sa3b = mk(63, 256, 512), sa3c = mk(67, 512, 1024);

  // ---- workspace bump allocator ----
  char* wp = (char*)d_ws;
  auto alloc = [&](size_t bytes) {
    void* p = wp;
    wp += (bytes + 255) & ~((size_t)255);
    return p;
  };
  float*    xyz   = (float*)alloc((size_t)B * N0 * 3 * 4);
  float*    l1x   = (float*)alloc((size_t)B * S1 * 3 * 4);
  float*    l2x   = (float*)alloc((size_t)B * S2 * 3 * 4);
  int*      fi1   = (int*)alloc((size_t)B * S1 * 4);
  int*      fi2   = (int*)alloc((size_t)B * S2 * 4);
  int*      gi1   = (int*)alloc((size_t)B * S1 * J1 * 4);
  int*      gi2   = (int*)alloc((size_t)B * S2 * J2 * 4);
  _Float16* l1p   = (_Float16*)alloc((size_t)B * S1 * 128 * 2);
  _Float16* l2p   = (_Float16*)alloc((size_t)B * S2 * 256 * 2);
  _Float16* l3p   = (_Float16*)alloc((size_t)B * 1024 * 2);
  _Float16* l2p2  = (_Float16*)alloc((size_t)B * S2 * 256 * 2);
  _Float16* l1p2  = (_Float16*)alloc((size_t)B * S1 * 128 * 2);
  int*      kidx  = (int*)alloc((size_t)B * N0 * 3 * 4);
  float*    kw    = (float*)alloc((size_t)B * N0 * 3 * 4);
  float*    logits = (float*)alloc((size_t)B * N0 * 50 * 4);
  _Float16* Whbuf = (_Float16*)alloc((size_t)1024 * 1280 * 2);   // max Coutp*Cinp
  _Float16* bufA  = (_Float16*)alloc((size_t)262144 * 160 * 2);  // ping
  _Float16* bufB  = (_Float16*)alloc((size_t)262144 * 160 * 2);  // pong

  auto blks = [](long long t) { return (unsigned)((t + 255) / 256); };
  auto gemm = [&](const _Float16* Xp, const Layer& L, int R, int bn,
                  _Float16* Yh, float* Yf) {
    long long wn = (long long)L.coutp * L.cinp;
    k_convert_w<<<blks(wn), 256, 0, stream>>>(L.W, L.cin, L.cout, L.cinp,
                                              L.coutp, Whbuf);
    dim3 grid((unsigned)(R / 64), (unsigned)(L.coutp / 32));
    k_gemm<<<grid, 128, 0, stream>>>(Xp, Whbuf, L.b, L.g, L.bb, Yh, Yf,
                                     R, L.cinp, L.cout, bn);
  };

  // ---- stage 0: coords ----
  k_transpose<<<blks((long long)B * N0), 256, 0, stream>>>(x, xyz, B, N0);

  // ---- SA1 ----
  k_fps<<<B, 256, 0, stream>>>(xyz, N0, S1, fi1);
  k_gather3<<<blks(B * S1), 256, 0, stream>>>(xyz, fi1, N0, S1, B * S1, l1x);
  k_query_ball<<<blks(B * S1), 256, 0, stream>>>(xyz, l1x, N0, S1, J1,
                                                 0.2f * 0.2f, B * S1, gi1);
  int R1 = B * S1 * J1;  // 262144
  k_group<<<blks(R1), 256, 0, stream>>>(xyz, l1x, gi1, (const _Float16*)nullptr,
                                        0, N0, S1, J1, R1, sa1a.cinp, bufA);
  gemm(bufA, sa1a, R1, 1, bufB, nullptr);
  gemm(bufB, sa1b, R1, 1, bufA, nullptr);
  gemm(bufA, sa1c, R1, 1, bufB, nullptr);
  k_maxpool<<<blks((long long)B * S1 * 128), 256, 0, stream>>>(bufB, B * S1, J1, 128, l1p);

  // ---- SA2 ----
  k_fps<<<B, 256, 0, stream>>>(l1x, S1, S2, fi2);
  k_gather3<<<blks(B * S2), 256, 0, stream>>>(l1x, fi2, S1, S2, B * S2, l2x);
  k_query_ball<<<blks(B * S2), 256, 0, stream>>>(l1x, l2x, S1, S2, J2,
                                                 0.4f * 0.4f, B * S2, gi2);
  int R2 = B * S2 * J2;  // 131072
  k_group<<<blks(R2), 256, 0, stream>>>(l1x, l2x, gi2, l1p, 128, S1, S2, J2,
                                        R2, sa2a.cinp, bufA);
  gemm(bufA, sa2a, R2, 1, bufB, nullptr);
  gemm(bufB, sa2b, R2, 1, bufA, nullptr);
  gemm(bufA, sa2c, R2, 1, bufB, nullptr);
  k_maxpool<<<blks((long long)B * S2 * 256), 256, 0, stream>>>(bufB, B * S2, J2, 256, l2p);

  // ---- SA3 (group all) ----
  int R3 = B * S2;  // 2048
  k_group_all<<<blks(R3), 256, 0, stream>>>(l2x, l2p, R3, sa3a.cinp, bufA);
  gemm(bufA, sa3a, R3, 1, bufB, nullptr);
  gemm(bufB, sa3b, R3, 1, bufA, nullptr);
  gemm(bufA, sa3c, R3, 1, bufB, nullptr);
  k_maxpool<<<blks((long long)B * 1024), 256, 0, stream>>>(bufB, B, S2, 1024, l3p);

  // ---- FP3 (S==1: broadcast) ----
  k_concat_fp3<<<blks((long long)R3 * 1280), 256, 0, stream>>>(l2p, l3p, R3, bufA);
  gemm(bufA, fp3a, R3, 1, bufB, nullptr);
  gemm(bufB, fp3b, R3, 1, l2p2, nullptr);

  // ---- FP2 ----
  int R4 = B * S1;  // 8192
  k_knn3<<<blks(R4), 256, 0, stream>>>(l1x, l2x, R4, S1, S2, kidx, kw);
  k_interp_concat<<<blks((long long)R4 * 384), 256, 0, stream>>>(
      l1p, 128, l2p2, 256, kidx, kw, R4, S2, bufA);
  gemm(bufA, fp2a, R4, 1, bufB, nullptr);
  gemm(bufB, fp2b, R4, 1, l1p2, nullptr);

  // ---- FP1 ----
  int R5 = B * N0;  // 65536
  k_knn3<<<blks(R5), 256, 0, stream>>>(xyz, l1x, R5, N0, S1, kidx, kw);
  k_interp_concat<<<blks((long long)R5 * 128), 256, 0, stream>>>(
      (const _Float16*)nullptr, 0, l1p2, 128, kidx, kw, R5, S1, bufA);
  gemm(bufA, fp1a, R5, 1, bufB, nullptr);
  gemm(bufB, fp1b, R5, 1, bufA, nullptr);
  gemm(bufA, fp1c, R5, 1, bufB, nullptr);

  // ---- head + log_softmax ----
  gemm(bufB, head1, R5, 1, bufA, nullptr);
  gemm(bufA, head2, R5, 0, nullptr, logits);
  k_logsoftmax<<<blks(R5), 256, 0, stream>>>(logits, (float*)d_out, R5, 50);
}
